// SparseExpertApplication_24068996727035
// MI455X (gfx1250) — compile-verified, matrix operand order checked by reading, not run on hardware
//
#include <hip/hip_runtime.h>
#include <math.h>

typedef __attribute__((ext_vector_type(16))) __bf16 v16bf;
typedef __attribute__((ext_vector_type(8)))  float  v8f;

#define BB 4096   // tokens
#define DD 1024   // model dim
#define HH 4096   // hidden dim
#define EE 8      // experts

#define BM 128    // workgroup M tile
#define BN 256    // workgroup N tile
#define BK 32     // K step (= WMMA K for bf16)
// 8 waves: 2 (M) x 4 (N), each wave owns a 64x64 tile = 4x4 WMMA accumulators.

// ---------- helpers ----------

// Truncating f32x2 -> packed bf16x2 in ONE v_perm_b32. Low 16 = a (even k), high = b (odd k).
__device__ __forceinline__ unsigned pack_bf16x2(float a, float b) {
  return __builtin_amdgcn_perm(__float_as_uint(b), __float_as_uint(a), 0x07060302u);
}

// RNE f32 -> bf16 (epilogue only, off the critical path)
__device__ __forceinline__ unsigned short f32_to_bf16(float a) {
  unsigned u = __float_as_uint(a);
  return (unsigned short)((u + 0x7FFFu + ((u >> 16) & 1u)) >> 16);
}

// WMMA 16-bit A/B fragment word position (ISA 7.12.2). Word packs k=2p,2p+1.
//   p0-3 -> lanes lo, v=p; p4-7 -> lanes hi, v=p-4; p8-11 -> lanes lo, v=4+; p12-15 -> lanes hi, v=4+
// LDS index v*32+lane => fragment reads are lane-stride-4B (conflict-free), mergeable to 2addr.
__device__ __forceinline__ int frag_word(int r, int p) {
  int v    = (p & 3) | ((p & 8) ? 4 : 0);
  int lane = r + ((p & 4) ? 16 : 0);
  return v * 32 + lane;
}

// tanh-form GELU: ~10 VALU (2 TRANS) instead of ~40 for the erff expansion.
// |error| vs exact erf-GELU ~1e-3 abs, far below bf16 GEMM noise.
__device__ __forceinline__ float gelu_fast(float x) {
  float t = 0.7978845608028654f * (x + 0.044715f * x * x * x);
  float a = __builtin_fabsf(t);
  float e = __expf(2.0f * a);                       // v_exp_f32 path
  float th = 1.0f - 2.0f * __builtin_amdgcn_rcpf(e + 1.0f);
  th = t < 0.0f ? -th : th;
  return 0.5f * x * (1.0f + th);
}

// ---------- setup kernels ----------

__global__ void k_zero_counts(int* counts) {
  if (threadIdx.x < EE) counts[threadIdx.x] = 0;
}

__global__ void k_zero_out(float* out) {
  size_t i = (size_t)blockIdx.x * blockDim.x + threadIdx.x;
  out[i] = 0.0f;
}

// Per-expert compacted token lists; dedupe the two slots (sum gates) so each
// (token,expert) appears once -> GEMM2 epilogue RMW on `out` rows is race-free.
__global__ void k_route(const int* __restrict__ eidx, const float* __restrict__ hw,
                        int* __restrict__ counts, int* __restrict__ list,
                        float* __restrict__ gatew) {
  int b = blockIdx.x * blockDim.x + threadIdx.x;
  if (b >= BB) return;
  int   e0 = eidx[b * 2 + 0], e1 = eidx[b * 2 + 1];
  float w0 = hw[b * 2 + 0],   w1 = hw[b * 2 + 1];
  if (e0 == e1) {
    int pos = atomicAdd(&counts[e0], 1);
    list[e0 * BB + pos]  = b;
    gatew[e0 * BB + pos] = w0 + w1;
  } else {
    int p0 = atomicAdd(&counts[e0], 1);
    list[e0 * BB + p0]  = b;
    gatew[e0 * BB + p0] = w0;
    int p1 = atomicAdd(&counts[e1], 1);
    list[e1 * BB + p1]  = b;
    gatew[e1 * BB + p1] = w1;
  }
}

// ---------- GEMM1: h = gelu(Xg @ W1[e] + b1[e]) ,  M=cnt, N=HH, K=DD ----------

__global__ __launch_bounds__(256) void k_gemm1(
    const float* __restrict__ x,       // [B, D]
    const float* __restrict__ w1,      // [E, D, H]
    const float* __restrict__ b1,      // [E, H]
    const int* __restrict__ counts,
    const int* __restrict__ list,      // [E, B]
    unsigned short* __restrict__ hbuf, // [B, H] bf16, compact rows per expert
    int e) {
  const int tid = threadIdx.x;
  const int lane = tid & 31, wid = tid >> 5;
  const int wave_m = wid >> 2, wave_n = wid & 3;

  const int cnt = counts[e];
  const int m_base = blockIdx.y * BM;
  if (m_base >= cnt) return;
  const int n_base = blockIdx.x * BN;

  __shared__ unsigned lA[8][256];   // 8  x (16x32 bf16 tile), fragment layout
  __shared__ unsigned lB[16][256];  // 16 x (32x16 bf16 tile)
  unsigned* lAf = &lA[0][0];
  unsigned* lBf = &lB[0][0];

  // A staging: 128 rows x 8 float4 (k-span 32) = 1024 float4 / 256 thr = 4 each.
  const float* aptr[4];
  int atgt[4];
  #pragma unroll
  for (int j = 0; j < 4; ++j) {
    int id = tid + 256 * j;
    int m = id >> 3, f = id & 7;               // row, float4 index along k
    int row = m_base + m;
    int crow = row < cnt ? row : (cnt - 1);
    int tok = list[e * BB + crow];
    aptr[j] = x + (size_t)tok * DD + 4 * f;
    atgt[j] = (m >> 4) * 256 + frag_word(m & 15, 2 * f);  // words p=2f and 2f+1 (+32)
  }

  // B staging: 16 p x 64 n-quads = 1024 quads / 256 thr = 4 each (2 float4 loads per quad).
  const float* wbase = w1 + (size_t)e * DD * HH;
  const float* bptr[4];
  int btgt[4];
  #pragma unroll
  for (int j = 0; j < 4; ++j) {
    int q = tid + 256 * j;
    int nq = q & 63, p = q >> 6;
    int n = nq * 4;
    bptr[j] = wbase + (size_t)(2 * p) * HH + (n_base + n);
    btgt[j] = (n >> 4) * 256 + frag_word(n & 15, p);      // 4 consecutive words
  }

  float4 aR[4], bR0[4], bR1[4];
  #pragma unroll
  for (int j = 0; j < 4; ++j) {
    aR[j]  = *(const float4*)(aptr[j]);
    bR0[j] = *(const float4*)(bptr[j]);
    bR1[j] = *(const float4*)(bptr[j] + HH);
  }

  v8f c[4][4] = {};

  for (int k0 = 0; k0 < DD; k0 += BK) {
    // pack prefetched tile into LDS (1 v_perm per word)
    #pragma unroll
    for (int j = 0; j < 4; ++j) {
      lAf[atgt[j]]      = pack_bf16x2(aR[j].x, aR[j].y);
      lAf[atgt[j] + 32] = pack_bf16x2(aR[j].z, aR[j].w);
      uint4 wv;
      wv.x = pack_bf16x2(bR0[j].x, bR1[j].x);
      wv.y = pack_bf16x2(bR0[j].y, bR1[j].y);
      wv.z = pack_bf16x2(bR0[j].z, bR1[j].z);
      wv.w = pack_bf16x2(bR0[j].w, bR1[j].w);
      *(uint4*)&lBf[btgt[j]] = wv;
    }
    __syncthreads();

    union { unsigned u[8]; v16bf v; } af[4], bfr[4];
    #pragma unroll
    for (int mt = 0; mt < 4; ++mt)
      #pragma unroll
      for (int v = 0; v < 8; ++v) af[mt].u[v] = lA[wave_m * 4 + mt][v * 32 + lane];
    #pragma unroll
    for (int nt = 0; nt < 4; ++nt)
      #pragma unroll
      for (int v = 0; v < 8; ++v) bfr[nt].u[v] = lB[wave_n * 4 + nt][v * 32 + lane];

    // prefetch next K tile while WMMAs run
    if (k0 + BK < DD) {
      #pragma unroll
      for (int j = 0; j < 4; ++j) {
        aR[j]  = *(const float4*)(aptr[j] + (k0 + BK));
        bR0[j] = *(const float4*)(bptr[j] + (size_t)(k0 + BK) * HH);
        bR1[j] = *(const float4*)(bptr[j] + (size_t)(k0 + BK) * HH + HH);
      }
    }

    #pragma unroll
    for (int mt = 0; mt < 4; ++mt)
      #pragma unroll
      for (int nt = 0; nt < 4; ++nt)
        c[mt][nt] = __builtin_amdgcn_wmma_f32_16x16x32_bf16(
            false, af[mt].v, false, bfr[nt].v, (short)0, c[mt][nt], false, false);
    __syncthreads();
  }

  // Epilogue: C layout = VGPR r: lanes0-15 -> (M=r, N=lane); lanes16-31 -> (M=8+r, N=lane-16)
  const int l16 = lane & 15, lhi = lane >> 4;
  #pragma unroll
  for (int nt = 0; nt < 4; ++nt) {
    int n = n_base + wave_n * 64 + nt * 16 + l16;
    float bias = b1[e * HH + n];
    #pragma unroll
    for (int mt = 0; mt < 4; ++mt) {
      #pragma unroll
      for (int r = 0; r < 8; ++r) {
        int m = m_base + wave_m * 64 + mt * 16 + r + 8 * lhi;
        if (m < cnt)
          hbuf[(size_t)m * HH + n] = f32_to_bf16(gelu_fast(c[mt][nt][r] + bias));
      }
    }
  }
}

// ---------- GEMM2: out[tok] += gate * (h @ W2[e] + b2[e]) ,  M=cnt, N=DD, K=HH ----------

__global__ __launch_bounds__(256) void k_gemm2(
    const unsigned short* __restrict__ hbuf, // [B, H] bf16 (pre-packed WMMA words)
    const float* __restrict__ w2,            // [E, H, D]
    const float* __restrict__ b2,            // [E, D]
    const int* __restrict__ counts,
    const int* __restrict__ list,            // [E, B]
    const float* __restrict__ gatew,         // [E, B]
    float* __restrict__ out,                 // [B, D]
    int e) {
  const int tid = threadIdx.x;
  const int lane = tid & 31, wid = tid >> 5;
  const int wave_m = wid >> 2, wave_n = wid & 3;

  const int cnt = counts[e];
  const int m_base = blockIdx.y * BM;
  if (m_base >= cnt) return;
  const int n_base = blockIdx.x * BN;

  __shared__ unsigned lA[8][256];
  __shared__ unsigned lB[16][256];
  unsigned* lAf = &lA[0][0];
  unsigned* lBf = &lB[0][0];

  // A staging: rows already bf16-packed -> pure uint4 copy. 128 rows x 4 word-quads
  // (16 words = 32 k) = 512 quads / 256 thr = 2 each.
  const unsigned* aptr[2];
  int atgt[2];
  #pragma unroll
  for (int j = 0; j < 2; ++j) {
    int q = tid + 256 * j;
    int m = q >> 2, fq = q & 3;
    int row = m_base + m;
    int crow = row < cnt ? row : (cnt - 1);
    aptr[j] = (const unsigned*)(hbuf + (size_t)crow * HH) + 4 * fq;
    atgt[j] = (m >> 4) * 256 + frag_word(m & 15, 4 * fq);  // words at +0,+32,+64,+96
  }

  const float* wbase = w2 + (size_t)e * HH * DD;
  const float* bptr[4];
  int btgt[4];
  #pragma unroll
  for (int j = 0; j < 4; ++j) {
    int q = tid + 256 * j;
    int nq = q & 63, p = q >> 6;
    int n = nq * 4;
    bptr[j] = wbase + (size_t)(2 * p) * DD + (n_base + n);
    btgt[j] = (n >> 4) * 256 + frag_word(n & 15, p);
  }

  uint4 aR[2];
  float4 bR0[4], bR1[4];
  #pragma unroll
  for (int j = 0; j < 2; ++j) aR[j] = *(const uint4*)(aptr[j]);
  #pragma unroll
  for (int j = 0; j < 4; ++j) {
    bR0[j] = *(const float4*)(bptr[j]);
    bR1[j] = *(const float4*)(bptr[j] + DD);
  }

  v8f c[4][4] = {};

  for (int k0 = 0; k0 < HH; k0 += BK) {
    #pragma unroll
    for (int j = 0; j < 2; ++j) {
      lAf[atgt[j]]      = aR[j].x;
      lAf[atgt[j] + 32] = aR[j].y;
      lAf[atgt[j] + 64] = aR[j].z;
      lAf[atgt[j] + 96] = aR[j].w;
    }
    #pragma unroll
    for (int j = 0; j < 4; ++j) {
      uint4 wv;
      wv.x = pack_bf16x2(bR0[j].x, bR1[j].x);
      wv.y = pack_bf16x2(bR0[j].y, bR1[j].y);
      wv.z = pack_bf16x2(bR0[j].z, bR1[j].z);
      wv.w = pack_bf16x2(bR0[j].w, bR1[j].w);
      *(uint4*)&lBf[btgt[j]] = wv;
    }
    __syncthreads();

    union { unsigned u[8]; v16bf v; } af[4], bfr[4];
    #pragma unroll
    for (int mt = 0; mt < 4; ++mt)
      #pragma unroll
      for (int v = 0; v < 8; ++v) af[mt].u[v] = lA[wave_m * 4 + mt][v * 32 + lane];
    #pragma unroll
    for (int nt = 0; nt < 4; ++nt)
      #pragma unroll
      for (int v = 0; v < 8; ++v) bfr[nt].u[v] = lB[wave_n * 4 + nt][v * 32 + lane];

    if (k0 + BK < HH) {
      #pragma unroll
      for (int j = 0; j < 2; ++j) aR[j] = *(const uint4*)(aptr[j] + ((k0 + BK) >> 1));
      #pragma unroll
      for (int j = 0; j < 4; ++j) {
        bR0[j] = *(const float4*)(bptr[j] + (size_t)(k0 + BK) * DD);
        bR1[j] = *(const float4*)(bptr[j] + (size_t)(k0 + BK) * DD + DD);
      }
    }

    #pragma unroll
    for (int mt = 0; mt < 4; ++mt)
      #pragma unroll
      for (int nt = 0; nt < 4; ++nt)
        c[mt][nt] = __builtin_amdgcn_wmma_f32_16x16x32_bf16(
            false, af[mt].v, false, bfr[nt].v, (short)0, c[mt][nt], false, false);
    __syncthreads();
  }

  // Epilogue: out[tok, n] += gate * (acc + b2). Rows unique per expert; expert
  // kernels are stream-serialized -> plain RMW is safe and deterministic.
  const int l16 = lane & 15, lhi = lane >> 4;
  float b2v[4];
  int ncol[4];
  #pragma unroll
  for (int nt = 0; nt < 4; ++nt) {
    ncol[nt] = n_base + wave_n * 64 + nt * 16 + l16;
    b2v[nt]  = b2[e * DD + ncol[nt]];
  }
  #pragma unroll
  for (int mt = 0; mt < 4; ++mt) {
    #pragma unroll
    for (int r = 0; r < 8; ++r) {
      int m = m_base + wave_m * 64 + mt * 16 + r + 8 * lhi;
      if (m < cnt) {
        int   tok = list[e * BB + m];
        float g   = gatew[e * BB + m];
        float* orow = out + (size_t)tok * DD;
        #pragma unroll
        for (int nt = 0; nt < 4; ++nt)
          orow[ncol[nt]] += g * (c[mt][nt][r] + b2v[nt]);
      }
    }
  }
}

// ---------- launch ----------

extern "C" void kernel_launch(void* const* d_in, const int* in_sizes, int n_in,
                              void* d_out, int out_size, void* d_ws, size_t ws_size,
                              hipStream_t stream) {
  const float* x    = (const float*)d_in[0];
  const int*   eidx = (const int*)d_in[1];
  const float* hw   = (const float*)d_in[2];
  const float* w1   = (const float*)d_in[3];
  const float* b1   = (const float*)d_in[4];
  const float* w2   = (const float*)d_in[5];
  const float* b2   = (const float*)d_in[6];
  float* out = (float*)d_out;

  char* ws = (char*)d_ws;
  int*   counts = (int*)ws;                                   // 32 B (padded to 256)
  int*   list   = (int*)(ws + 256);                           // E*B*4 = 128 KB
  float* gatew  = (float*)(ws + 256 + EE * BB * 4);           // E*B*4 = 128 KB
  unsigned short* hbuf =
      (unsigned short*)(ws + 256 + 2 * (size_t)EE * BB * 4);  // B*H*2 = 32 MB

  k_zero_counts<<<1, 64, 0, stream>>>(counts);
  k_zero_out<<<(BB * DD) / 256, 256, 0, stream>>>(out);
  k_route<<<BB / 256, 256, 0, stream>>>(eidx, hw, counts, list, gatew);

  for (int e = 0; e < EE; ++e) {
    k_gemm1<<<dim3(HH / BN, BB / BM), 256, 0, stream>>>(x, w1, b1, counts, list, hbuf, e);
    k_gemm2<<<dim3(DD / BN, BB / BM), 256, 0, stream>>>(hbuf, w2, b2, counts, list, gatew, out, e);
  }
}
